// VoxelToGaussianLifter_14568529068113
// MI455X (gfx1250) — matrix operand order repaired and checked by reading, not compiled, not run on hardware
//
#include <hip/hip_runtime.h>
#include <hip/hip_bf16.h>

typedef __attribute__((ext_vector_type(16))) _Float16 v16h;
typedef __attribute__((ext_vector_type(8)))  float    v8f;

#define NVOX   20000
#define NGAUSS 25600
#define PROPD  28
#define EMBEDD 128
#define BB     2
#define PP     2000000
#define TILES_PER_B (NVOX / 16)      // 1250
#define NTILES (BB * TILES_PER_B)    // 2500

// ---------------- helpers ----------------

__device__ inline v8f wmma_f16(v16h a, v16h b, v8f c) {
  return __builtin_amdgcn_wmma_f32_16x16x32_f16(
      /*neg_a=*/false, a, /*neg_b=*/false, b,
      /*c_mod=*/(short)0, c, /*reuse_a=*/false, /*reuse_b=*/false);
}

// A-fragment (16x32 f16): lane holds row M=lane%16; half=lane/16.
// element pair j covers K0 = (j<4 ? 2j : 16+2(j-4)) + 8*half  -> dword (j<4?j:j+4)+4*half
__device__ inline v16h load_a_dwords(const unsigned* __restrict__ rowbase, int half, int dwOff) {
  union { unsigned u[8]; v16h h; } r;
#pragma unroll
  for (int j = 0; j < 8; ++j) {
    int dw = dwOff + ((j < 4) ? j : (j + 4)) + 4 * half;
    r.u[j] = rowbase[dw];
  }
  return r.h;
}

// B-fragment pre-packed as [tile][j][lane] dwords -> fully coalesced loads
__device__ inline v16h load_b_packed(const unsigned* __restrict__ base, int lane) {
  union { unsigned u[8]; v16h h; } r;
#pragma unroll
  for (int j = 0; j < 8; ++j) r.u[j] = base[j * 32 + lane];
  return r.h;
}

// ---------------- kernels ----------------

__global__ void zero_acc_kernel(float* __restrict__ acc, int n) {
  int i = blockIdx.x * blockDim.x + threadIdx.x;
  if (i < n) acc[i] = 0.0f;
}

// Single-pass scatter: cnt, sum xyz, sumsq xyz, sum i, sumsq i (9 slots, stride 12)
__global__ void accum_kernel(const float4* __restrict__ pts,
                             const int* __restrict__ inv,
                             float* __restrict__ acc) {
  int i = blockIdx.x * blockDim.x + threadIdx.x;
  if (i >= BB * PP) return;
  float4 p = pts[i];
  int b = i / PP;
  int v = inv[i];
  float* a = acc + (size_t)(b * NVOX + v) * 12;
  float vals[9] = {1.0f, p.x, p.y, p.z, p.x * p.x, p.y * p.y, p.z * p.z,
                   p.w, p.w * p.w};
#pragma unroll
  for (int k = 0; k < 9; ++k)
    __hip_atomic_fetch_add(&a[k], vals[k], __ATOMIC_RELAXED, __HIP_MEMORY_SCOPE_AGENT);
}

// Per-voxel stats -> f16, K padded to 32 with zeros (WMMA-ready rows)
__global__ void stats_kernel(const float* __restrict__ acc,
                             _Float16* __restrict__ statsH) {
  int i = blockIdx.x * blockDim.x + threadIdx.x;
  if (i >= BB * NVOX) return;
  const float* a = acc + (size_t)i * 12;
  float cnt = a[0];
  float cc  = fmaxf(cnt, 1.0f);
  float mx = a[1] / cc, my = a[2] / cc, mz = a[3] / cc;
  float sx = sqrtf(fmaxf(a[4] / cc - mx * mx, 0.0f));
  float sy = sqrtf(fmaxf(a[5] / cc - my * my, 0.0f));
  float sz = sqrtf(fmaxf(a[6] / cc - mz * mz, 0.0f));
  float mi = a[7] / cc;
  float si = sqrtf(fmaxf(a[8] / cc - mi * mi, 0.0f));
  float s[9] = {mx, my, mz, sx, sy, sz, fminf(cnt, 10.0f) * 0.1f, mi, si};
  _Float16* o = statsH + (size_t)i * 32;
#pragma unroll
  for (int k = 0; k < 32; ++k) o[k] = (k < 9) ? (_Float16)s[k] : (_Float16)0.0f;
}

// fetch one weight element for B-fragment packing
__device__ inline float fetch_w(int t, int K, int lane16,
                                const float* __restrict__ w1f,
                                const float* __restrict__ w2f,
                                const float* __restrict__ w1p,
                                const float* __restrict__ w2p) {
  if (t < 8) {                       // w1f: 9x128, n-tile = t, kc=0
    int N = t * 16 + lane16;
    return (K < 9) ? w1f[K * 128 + N] : 0.0f;
  } else if (t < 40) {               // w2f: 128x128, t-8 = n*4+kc
    int tt = t - 8; int n = tt >> 2; int kc = tt & 3;
    return w2f[(kc * 32 + K) * 128 + (n * 16 + lane16)];
  } else if (t < 44) {               // w1p: 9x64
    int n = t - 40; int N = n * 16 + lane16;
    return (K < 9) ? w1p[K * 64 + N] : 0.0f;
  } else {                           // w2p: 64x28, t-44 = n*2+kc
    int tt = t - 44; int n = tt >> 1; int kc = tt & 1;
    int N = n * 16 + lane16;
    return (N < PROPD) ? w2p[(kc * 32 + K) * PROPD + N] : 0.0f;
  }
}

// Pack all four weight matrices into exact B-fragment VGPR layout (48 tiles x 256 dwords)
__global__ void pack_weights_kernel(const float* __restrict__ w1f,
                                    const float* __restrict__ w2f,
                                    const float* __restrict__ w1p,
                                    const float* __restrict__ w2p,
                                    unsigned* __restrict__ packedB) {
  int idx = blockIdx.x * blockDim.x + threadIdx.x;
  if (idx >= 48 * 256) return;
  int lane = idx & 31;
  int j = (idx >> 5) & 7;
  int t = idx >> 8;
  int lane16 = lane & 15;
  int K0 = ((j < 4) ? 2 * j : 2 * j + 8) + 8 * (lane >> 4);
  float v0 = fetch_w(t, K0,     lane16, w1f, w2f, w1p, w2p);
  float v1 = fetch_w(t, K0 + 1, lane16, w1f, w2f, w1p, w2p);
  union { _Float16 h[2]; unsigned u; } x;
  x.h[0] = (_Float16)v0; x.h[1] = (_Float16)v1;
  packedB[t * 256 + j * 32 + lane] = x.u;
}

// WMMA MLPs: one wave = one 16-voxel tile; both MLPs, LDS staging between layers.
__global__ __launch_bounds__(128)
void mlp_wmma_kernel(const unsigned* __restrict__ statsU,
                     const unsigned* __restrict__ packedB,
                     const float* __restrict__ b1f, const float* __restrict__ b2f,
                     const float* __restrict__ b1p, const float* __restrict__ b2p,
                     float* __restrict__ out_p, float* __restrict__ out_f) {
  __shared__ _Float16 lds[4][16 * 128];   // 16 KB: per-wave 16x128 f16 staging
  int lane = threadIdx.x & 31;
  int w    = threadIdx.x >> 5;
  int tileG = blockIdx.x * 4 + w;               // grid sized exactly: 625*4 = 2500
  int b    = tileG / TILES_PER_B;
  int tile = tileG % TILES_PER_B;
  int l16  = lane & 15, half = lane >> 4;

  // A0: stats rows (K padded to 32 with zeros)
  int statRow = b * NVOX + tile * 16 + l16;
  const unsigned* arow = statsU + (size_t)statRow * 16;
  v16h a0 = load_a_dwords(arow, half, 0);

  // ---- features layer 1: 9 -> 128, relu, stage to LDS as f16 ----
#pragma unroll
  for (int n = 0; n < 8; ++n) {
    v16h bf = load_b_packed(packedB + n * 256, lane);
    v8f c = {};
    c = wmma_f16(a0, bf, c);
    float bias = b1f[n * 16 + l16];
#pragma unroll
    for (int r = 0; r < 8; ++r) {
      int M = r + 8 * half;
      lds[w][M * 128 + n * 16 + l16] = (_Float16)fmaxf(c[r] + bias, 0.0f);
    }
  }
  __syncthreads();

  // A-fragments for layer 2 (row M = l16, 64 dwords per row)
  const unsigned* frow = (const unsigned*)&lds[w][l16 * 128];
  v16h af[4];
#pragma unroll
  for (int kc = 0; kc < 4; ++kc) af[kc] = load_a_dwords(frow, half, kc * 16);

  // ---- features layer 2: 128 -> 128 ----
#pragma unroll
  for (int n = 0; n < 8; ++n) {
    v8f c = {};
#pragma unroll
    for (int kc = 0; kc < 4; ++kc) {
      v16h bf = load_b_packed(packedB + (8 + n * 4 + kc) * 256, lane);
      c = wmma_f16(af[kc], bf, c);
    }
    float bias = b2f[n * 16 + l16];
    int col = n * 16 + l16;
#pragma unroll
    for (int r = 0; r < 8; ++r) {
      int g = b * NGAUSS + tile * 16 + r + 8 * half;
      out_f[(size_t)g * EMBEDD + col] = c[r] + bias;
    }
  }
  __syncthreads();

  // ---- props layer 1: 9 -> 64, relu, stage (stride 64) ----
#pragma unroll
  for (int n = 0; n < 4; ++n) {
    v16h bf = load_b_packed(packedB + (40 + n) * 256, lane);
    v8f c = {};
    c = wmma_f16(a0, bf, c);
    float bias = b1p[n * 16 + l16];
#pragma unroll
    for (int r = 0; r < 8; ++r) {
      int M = r + 8 * half;
      lds[w][M * 64 + n * 16 + l16] = (_Float16)fmaxf(c[r] + bias, 0.0f);
    }
  }
  __syncthreads();

  const unsigned* prow = (const unsigned*)&lds[w][l16 * 64];  // 32 dwords per row
  v16h ap[2];
#pragma unroll
  for (int kc = 0; kc < 2; ++kc) ap[kc] = load_a_dwords(prow, half, kc * 16);

  // ---- props layer 2: 64 -> 28 (cols 28..31 discarded) ----
#pragma unroll
  for (int n = 0; n < 2; ++n) {
    v8f c = {};
#pragma unroll
    for (int kc = 0; kc < 2; ++kc) {
      v16h bf = load_b_packed(packedB + (44 + n * 2 + kc) * 256, lane);
      c = wmma_f16(ap[kc], bf, c);
    }
    int col = n * 16 + l16;
    if (col < PROPD) {
      float bias = b2p[col];
#pragma unroll
      for (int r = 0; r < 8; ++r) {
        int g = b * NGAUSS + tile * 16 + r + 8 * half;
        out_p[(size_t)g * PROPD + col] = c[r] + bias;
      }
    }
  }
}

// Fill padded rows [NVOX, NGAUSS) with defaults
__global__ void pad_kernel(const float* __restrict__ def_p,
                           const float* __restrict__ def_f,
                           float* __restrict__ out_p, float* __restrict__ out_f) {
  const int padRows = NGAUSS - NVOX;              // 5600
  const int totalP = BB * padRows * PROPD;        // 313600
  const int totalF = BB * padRows * EMBEDD;       // 1433600
  int i = blockIdx.x * blockDim.x + threadIdx.x;
  if (i < totalP) {
    int b = i / (padRows * PROPD);
    int rem = i % (padRows * PROPD);
    int row = NVOX + rem / PROPD, col = rem % PROPD;
    out_p[((size_t)b * NGAUSS + row) * PROPD + col] = def_p[col];
  } else if (i < totalP + totalF) {
    int k = i - totalP;
    int b = k / (padRows * EMBEDD);
    int rem = k % (padRows * EMBEDD);
    int row = NVOX + rem / EMBEDD, col = rem % EMBEDD;
    out_f[((size_t)b * NGAUSS + row) * EMBEDD + col] = def_f[col];
  }
}

// ---------------- launch ----------------

extern "C" void kernel_launch(void* const* d_in, const int* in_sizes, int n_in,
                              void* d_out, int out_size, void* d_ws, size_t ws_size,
                              hipStream_t stream) {
  const float4* points = (const float4*)d_in[0];
  const int*    inv    = (const int*)d_in[1];
  const float*  def_p  = (const float*)d_in[2];
  const float*  def_f  = (const float*)d_in[3];
  const float*  w1f    = (const float*)d_in[4];
  const float*  b1f    = (const float*)d_in[5];
  const float*  w2f    = (const float*)d_in[6];
  const float*  b2f    = (const float*)d_in[7];
  const float*  w1p    = (const float*)d_in[8];
  const float*  b1p    = (const float*)d_in[9];
  const float*  w2p    = (const float*)d_in[10];
  const float*  b2p    = (const float*)d_in[11];

  // workspace layout
  float*     acc     = (float*)d_ws;                                    // 2*20000*12 f32 = 1.92 MB
  _Float16*  statsH  = (_Float16*)((char*)d_ws + (size_t)BB * NVOX * 12 * 4);  // 2*20000*32 f16 = 2.56 MB
  unsigned*  packedB = (unsigned*)((char*)statsH + (size_t)BB * NVOX * 32 * 2); // 48*256 u32

  float* out_p = (float*)d_out;                                   // (B, NG, 28)
  float* out_f = out_p + (size_t)BB * NGAUSS * PROPD;             // (B, NG, 128)

  const int accN = BB * NVOX * 12;
  zero_acc_kernel<<<(accN + 255) / 256, 256, 0, stream>>>(acc, accN);
  pack_weights_kernel<<<48, 256, 0, stream>>>(w1f, w2f, w1p, w2p, packedB);
  accum_kernel<<<(BB * PP + 255) / 256, 256, 0, stream>>>(points, inv, acc);
  stats_kernel<<<(BB * NVOX + 255) / 256, 256, 0, stream>>>(acc, statsH);
  mlp_wmma_kernel<<<NTILES / 4, 128, 0, stream>>>((const unsigned*)statsH, packedB,
                                                  b1f, b2f, b1p, b2p, out_p, out_f);
  const int padTotal = BB * (NGAUSS - NVOX) * (PROPD + EMBEDD);
  pad_kernel<<<(padTotal + 255) / 256, 256, 0, stream>>>(def_p, def_f, out_p, out_f);
}
